// LSTMModel_64991445123112
// MI455X (gfx1250) — matrix operand order repaired and checked by reading, not compiled, not run on hardware
//
#include <hip/hip_runtime.h>
#include <hip/hip_bf16.h>
#include <stdint.h>

// Problem constants (match reference)
#define BATCH 64
#define SEQ   512
#define IN0   128
#define HID   256
#define GATES 1024   // 4*HID
#define IN1   512    // 2*HID

typedef __attribute__((ext_vector_type(16))) __bf16       bf16x16;
typedef __attribute__((ext_vector_type(8)))  float        f32x8;
typedef __attribute__((ext_vector_type(4)))  unsigned int u32x4;
typedef __attribute__((ext_vector_type(4)))  int          i32x4;

union FragB { bf16x16 v; u32x4 q[2]; };                 // 32B: A or B operand per lane
union FragC { f32x8   v; u32x4 q[2]; float f[8]; };     // 32B: C/D accumulator per lane

#if __has_builtin(__builtin_amdgcn_global_load_async_to_lds_b128)
#define USE_ASYNC_LDS 1
typedef __attribute__((address_space(1))) i32x4* gbl_v4i_ptr;   // global int4*
typedef __attribute__((address_space(3))) i32x4* lds_v4i_ptr;   // shared int4*
#else
#define USE_ASYNC_LDS 0
#endif

__device__ __forceinline__ unsigned short f2bf(float x) {
    uint32_t u = __builtin_bit_cast(uint32_t, x);
    uint32_t r = u + 0x7fffu + ((u >> 16) & 1u);        // round-to-nearest-even
    return (unsigned short)(r >> 16);
}
__device__ __forceinline__ float bf2f(unsigned short h) {
    uint32_t u = ((uint32_t)h) << 16;
    return __builtin_bit_cast(float, u);
}
__device__ __forceinline__ float fast_sig(float x) {
    return __builtin_amdgcn_rcpf(1.0f + __expf(-x));
}
__device__ __forceinline__ float fast_tanh(float x) {
    float e = __expf(2.0f * x);                          // +inf -> 1, 0 -> -1 handled
    return 1.0f - 2.0f * __builtin_amdgcn_rcpf(e + 1.0f);
}

// ---------------- fp32 -> bf16 elementwise convert ----------------
__global__ void cvt_f32_bf16(const float* __restrict__ in,
                             unsigned short* __restrict__ out, int n) {
    int i = blockIdx.x * blockDim.x + threadIdx.x;
    if (i < n) out[i] = f2bf(in[i]);
}

// ---------------- Input projection GEMM: xg = A @ W^T + bias ----------------
// A: bf16 [B][T][K] row-major.  W: bf16 [2][GATES][K].  bias: fp32 [2][GATES].
// Output stored as raw WMMA C fragments: [2][T][4 btile][64 ntile][32 lane][8] fp32.
// One WG per M-tile (16 batch rows at one t); A tile staged once in LDS
// (async global->LDS when available), 16 waves cover all 64 N-tiles.
__global__ __launch_bounds__(512)
void xg_gemm(const unsigned short* __restrict__ A,
             const unsigned short* __restrict__ W,
             const float* __restrict__ bias,
             float* __restrict__ xgf,
             int K)
{
    const int d    = blockIdx.y;
    const int mt   = blockIdx.x;          // M-tile: t = mt>>2, batch-tile = mt&3
    const int t    = mt >> 2;
    const int bt   = mt & 3;
    const int wave = threadIdx.x >> 5;    // 16 waves: N-tiles 4w..4w+3
    const int lane = threadIdx.x & 31;
    const int half = lane >> 4;
    const int l    = lane & 15;

    const unsigned short* Wd    = W + (size_t)d * GATES * K;
    const float*          biasd = bias + d * GATES;

    __shared__ __align__(16) unsigned short atile[16 * IN1];   // 16KB max (K<=512)

    // Stage the 16xK bf16 A tile into LDS cooperatively.
    const int rowc   = K / 8;             // 16B chunks per row
    const int chunks = 16 * rowc;
    for (int cidx = threadIdx.x; cidx < chunks; cidx += blockDim.x) {
        const int row = cidx / rowc;
        const int off = (cidx - row * rowc) * 8;
        const unsigned short* src = A + ((size_t)(bt * 16 + row) * SEQ + t) * K + off;
#if USE_ASYNC_LDS
        __builtin_amdgcn_global_load_async_to_lds_b128(
            (gbl_v4i_ptr)src,
            (lds_v4i_ptr)(atile + row * K + off),
            0, 0);
#else
        *(u32x4*)(atile + row * K + off) = *(const u32x4*)src;
#endif
    }
#if USE_ASYNC_LDS
#if __has_builtin(__builtin_amdgcn_s_wait_asynccnt)
    __builtin_amdgcn_s_wait_asynccnt(0);
#else
    asm volatile("s_wait_asynccnt 0" ::: "memory");
#endif
#endif
    __syncthreads();

    f32x8 acc[4];
    #pragma unroll
    for (int q = 0; q < 4; q++) acc[q] = (f32x8){0,0,0,0,0,0,0,0};

    const int n0 = wave * 4;
    for (int k0 = 0; k0 < K; k0 += 32) {
        FragB a;
        a.q[0] = *(const u32x4*)(atile + l * K + k0 + half * 8);       // K k0..k0+7 / +8..15
        a.q[1] = *(const u32x4*)(atile + l * K + k0 + 16 + half * 8);  // K +16..23 / +24..31
        #pragma unroll
        for (int q = 0; q < 4; q++) {
            const int n = n0 + q;
            const unsigned short* wrow = Wd + (size_t)(n * 16 + l) * K + k0 + half * 8;
            FragB b;
            b.q[0] = *(const u32x4*)(wrow);
            b.q[1] = *(const u32x4*)(wrow + 16);
            acc[q] = __builtin_amdgcn_wmma_f32_16x16x32_bf16(
                         false, a.v, false, b.v, (short)0, acc[q], false, false);
        }
    }
    #pragma unroll
    for (int q = 0; q < 4; q++) {
        const int n = n0 + q;
        const float bv = biasd[n * 16 + l];                    // col = n*16 + (lane&15)
        FragC c; c.v = acc[q];
        #pragma unroll
        for (int v = 0; v < 8; v++) c.f[v] += bv;
        size_t off = ((((size_t)d * SEQ + t) * 4 + bt) * 64 + n) * 32 + lane;
        u32x4* dst = (u32x4*)(xgf + off * 8);
        dst[0] = c.q[0]; dst[1] = c.q[1];
    }
}

// ---------------- Persistent LSTM scan (one layer, both dirs) ----------------
// Grid (4 batch-tiles, 2 dirs), 512 threads = 16 waves. Wave j owns h-slice
// [j*16, j*16+16) and gate tiles n = q*16+j (q = i,f,g,o). c stays in regs.
__global__ __launch_bounds__(512)
void lstm_scan(const float* __restrict__ xgf,            // [2][T][4][64][32][8]
               const unsigned short* __restrict__ Whh,   // bf16 [2][GATES][HID]
               unsigned short* __restrict__ out,         // bf16 [B][T][2H]
               int Tlen)
{
    const int bt   = blockIdx.x;
    const int d    = blockIdx.y;
    const int wave = threadIdx.x >> 5;   // h-subtile j
    const int lane = threadIdx.x & 31;
    const int half = lane >> 4;
    const int l    = lane & 15;

    __shared__ __align__(16) unsigned short hbuf[16 * HID];  // bf16 h, 8KB
    for (int i = threadIdx.x; i < 16 * HID; i += blockDim.x) hbuf[i] = 0;
    __syncthreads();

    const unsigned short* Wd = Whh + (size_t)d * GATES * HID;
    f32x8 c = (f32x8){0,0,0,0,0,0,0,0};

    for (int s = 0; s < Tlen; ++s) {
        const int t = d ? (Tlen - 1 - s) : s;

        // xg fragments (global, streamed from HBM) -- issue loads as early as
        // possible; independent of the LDS h state.
        FragC cc[4];
        #pragma unroll
        for (int q = 0; q < 4; q++) {
            const int n = q * 16 + wave;
            size_t off = ((((size_t)d * Tlen + t) * 4 + bt) * 64 + n) * 32 + lane;
            const u32x4* src = (const u32x4*)(xgf + off * 8);
            cc[q].q[0] = src[0]; cc[q].q[1] = src[1];
        }
        // prefetch next step's xg fragments (hides HBM latency on the
        // sequential critical path; emits global_prefetch_b8)
        if (s + 1 < Tlen) {
            const int tn = d ? (t - 1) : (t + 1);
            #pragma unroll
            for (int q = 0; q < 4; q++) {
                const int n = q * 16 + wave;
                size_t offn = ((((size_t)d * Tlen + tn) * 4 + bt) * 64 + n) * 32 + lane;
                __builtin_prefetch(xgf + offn * 8, 0, 1);
            }
        }

        // A fragments: h(s-1) from LDS, all K=256 (8 k-tiles)
        FragB a[8];
        #pragma unroll
        for (int k = 0; k < 8; k++) {
            const int k0 = k * 32;
            a[k].q[0] = *(const u32x4*)(hbuf + l * HID + k0 + half * 8);
            a[k].q[1] = *(const u32x4*)(hbuf + l * HID + k0 + 16 + half * 8);
        }
        __syncthreads();   // all reads of h(s-1) done before anyone writes h(s)

        f32x8 g[4];
        #pragma unroll
        for (int q = 0; q < 4; q++) {
            const int n = q * 16 + wave;
            f32x8 accv = cc[q].v;          // acc init = x@Wih^T + bias fragment
            #pragma unroll
            for (int k = 0; k < 8; k++) {
                const int k0 = k * 32;
                const unsigned short* wrow =
                    Wd + (size_t)(n * 16 + l) * HID + k0 + half * 8;
                FragB b;
                b.q[0] = *(const u32x4*)(wrow);
                b.q[1] = *(const u32x4*)(wrow + 16);
                accv = __builtin_amdgcn_wmma_f32_16x16x32_bf16(
                           false, a[k].v, false, b.v, (short)0, accv, false, false);
            }
            g[q] = accv;
        }

        // Gate nonlinearities + state update (fp32, elementwise on C layout)
        FragC gi, gf, gg, go, cold, cnew, hv;
        gi.v = g[0]; gf.v = g[1]; gg.v = g[2]; go.v = g[3]; cold.v = c;
        #pragma unroll
        for (int v = 0; v < 8; v++) {
            const float iv = fast_sig(gi.f[v]);
            const float fv = fast_sig(gf.f[v]);
            const float gv = fast_tanh(gg.f[v]);
            const float ov = fast_sig(go.f[v]);
            const float cv = fv * cold.f[v] + iv * gv;
            cnew.f[v] = cv;
            hv.f[v]   = ov * fast_tanh(cv);
        }
        c = cnew.v;

        // write h(s) to LDS (next step's A) and to the layer output
        const int col = wave * 16 + l;
        #pragma unroll
        for (int v = 0; v < 8; v++) {
            const int row = v + 8 * half;            // C layout: M = v + 8*half
            const unsigned short hb = f2bf(hv.f[v]);
            hbuf[row * HID + col] = hb;
            const int b = bt * 16 + row;
            out[((size_t)b * Tlen + t) * IN1 + d * HID + col] = hb;
        }
        __syncthreads();   // h(s) visible before next step's reads
    }
}

// ---------------- Final FC on t = T-1 ----------------
__global__ void fc_kernel(const unsigned short* __restrict__ h,  // bf16 [B][T][2H]
                          const float* __restrict__ fw,
                          const float* __restrict__ fb,
                          float* __restrict__ outp, int Tlen)
{
    const int b = blockIdx.x * blockDim.x + threadIdx.x;
    if (b >= BATCH) return;
    const unsigned short* row = h + ((size_t)b * Tlen + (Tlen - 1)) * IN1;
    float s = 0.f;
    for (int k = 0; k < IN1; k++) s += bf2f(row[k]) * fw[k];
    outp[b] = s + fb[0];
}

extern "C" void kernel_launch(void* const* d_in, const int* in_sizes, int n_in,
                              void* d_out, int out_size, void* d_ws, size_t ws_size,
                              hipStream_t stream) {
    const float* x      = (const float*)d_in[0];
    const float* w_ih_0 = (const float*)d_in[1];
    const float* w_hh_0 = (const float*)d_in[2];
    const float* b_0    = (const float*)d_in[3];
    const float* w_ih_1 = (const float*)d_in[4];
    const float* w_hh_1 = (const float*)d_in[5];
    const float* b_1    = (const float*)d_in[6];
    const float* fc_w   = (const float*)d_in[7];
    const float* fc_b   = (const float*)d_in[8];
    (void)in_sizes; (void)n_in; (void)out_size; (void)ws_size;

    // Workspace carve (256B-aligned chunks)
    char* ws = (char*)d_ws;
    size_t off = 0;
    auto take = [&](size_t bytes) { void* p = ws + off; off += (bytes + 255) & ~size_t(255); return p; };

    float*          xgf   = (float*)         take((size_t)2 * SEQ * 4 * 64 * 32 * 8 * sizeof(float)); // 256MB
    unsigned short* xbf   = (unsigned short*)take((size_t)BATCH * SEQ * IN0 * 2);                      // 8MB
    unsigned short* out0  = (unsigned short*)take((size_t)BATCH * SEQ * IN1 * 2);                      // 32MB
    unsigned short* out1  = (unsigned short*)take((size_t)BATCH * SEQ * IN1 * 2);                      // 32MB
    unsigned short* wih0b = (unsigned short*)take((size_t)2 * GATES * IN0 * 2);
    unsigned short* whh0b = (unsigned short*)take((size_t)2 * GATES * HID * 2);
    unsigned short* wih1b = (unsigned short*)take((size_t)2 * GATES * IN1 * 2);
    unsigned short* whh1b = (unsigned short*)take((size_t)2 * GATES * HID * 2);

    auto cvt = [&](const float* src, unsigned short* dst, int n) {
        cvt_f32_bf16<<<(n + 255) / 256, 256, 0, stream>>>(src, dst, n);
    };
    cvt(x,      xbf,   BATCH * SEQ * IN0);
    cvt(w_ih_0, wih0b, 2 * GATES * IN0);
    cvt(w_hh_0, whh0b, 2 * GATES * HID);
    cvt(w_ih_1, wih1b, 2 * GATES * IN1);
    cvt(w_hh_1, whh1b, 2 * GATES * HID);

    // Layer 0
    xg_gemm  <<<dim3(2048, 2), 512, 0, stream>>>(xbf,  wih0b, b_0, xgf, IN0);
    lstm_scan<<<dim3(4, 2),    512, 0, stream>>>(xgf, whh0b, out0, SEQ);
    // Layer 1 (reuse xgf)
    xg_gemm  <<<dim3(2048, 2), 512, 0, stream>>>(out0, wih1b, b_1, xgf, IN1);
    lstm_scan<<<dim3(4, 2),    512, 0, stream>>>(xgf, whh1b, out1, SEQ);
    // FC head
    fc_kernel<<<1, 64, 0, stream>>>(out1, fc_w, fc_b, (float*)d_out, SEQ);
}
